// SBOLayers_82583631168159
// MI455X (gfx1250) — matrix operand order repaired and checked by reading, not compiled, not run on hardware
//
#include <hip/hip_runtime.h>
#include <hip/hip_bf16.h>

typedef __bf16 v16bf __attribute__((ext_vector_type(16)));
typedef float  v8f   __attribute__((ext_vector_type(8)));

#define Bsz 8
#define Ssz 512
#define Hsz 768
#define H3  2304
#define Vsz 30522
#define Psz 512

__device__ __forceinline__ __bf16 f2bf(float f) {
    unsigned u = __builtin_bit_cast(unsigned, f);
    unsigned r = (u + 0x7FFFu + ((u >> 16) & 1u)) >> 16;
    return __builtin_bit_cast(__bf16, (unsigned short)r);
}

// ---------------------------------------------------------------------------
// Kernel 1: build masked concat [4096, 2304] in bf16
// ---------------------------------------------------------------------------
__global__ __launch_bounds__(256)
void build_concat_k(const float* __restrict__ x, const int* __restrict__ left,
                    const int* __restrict__ right, const unsigned char* __restrict__ mask,
                    const float* __restrict__ pos, __bf16* __restrict__ out) {
    const int row = blockIdx.x;              // b*S + s
    const int b = row >> 9, s = row & 511;
    const int li = left[row], ri = right[row];
    int rel = s - li;
    rel = rel < 0 ? 0 : (rel > Psz - 1 ? Psz - 1 : rel);
    const bool mk = mask[row] != 0;
    const float* xl = x + ((size_t)b * Ssz + li) * Hsz;
    const float* xr = x + ((size_t)b * Ssz + ri) * Hsz;
    const float* pp = pos + (size_t)rel * Hsz;
    __bf16* o = out + (size_t)row * H3;
    for (int i = threadIdx.x; i < Hsz; i += 256) {
        float a = mk ? xl[i] : -100.0f;
        float c = mk ? xr[i] : -100.0f;
        float d = mk ? pp[i] : -100.0f;
        o[i]            = f2bf(a);
        o[Hsz + i]      = f2bf(c);
        o[2 * Hsz + i]  = f2bf(d);
    }
}

// ---------------------------------------------------------------------------
// Kernel 2: bf16 WMMA GEMM  C[M,N] = A[M,K](bf16) * Bw[K,N](f32->bf16) + bias
// Block tile 128x128, BK=32, double-buffered LDS.
//   A tile: async global->LDS copies (ASYNCcnt), bf16 direct.
//   B tile: f32 global loads prefetched into regs, cvt+ds_store after compute.
// 8 waves (2x4). Wave tile 64x32 = 4x2 WMMA fragments.
// ---------------------------------------------------------------------------
#define LDA 40   // bf16 elems; 80B row stride, 16B aligned for b128 async stores
#define LDB 34

__device__ __forceinline__ void stage_A_async(const __bf16* Ak, unsigned ldsBase,
                                              int tid, int K) {
    // 128 rows x 64B payload = 512 segments of 16B; 2 per thread
#pragma unroll
    for (int q = 0; q < 2; ++q) {
        int seg  = tid + q * 256;
        int row  = seg >> 2, part = seg & 3;
        unsigned loff = ldsBase + (unsigned)(row * (LDA * 2) + part * 16);
        unsigned goff = (unsigned)(row * K * 2 + part * 16);
        asm volatile("global_load_async_to_lds_b128 %0, %1, %2"
                     :: "v"(loff), "v"(goff), "s"(Ak)
                     : "memory");
    }
}

__device__ __forceinline__ void load_B_regs(float* r, const float* __restrict__ Bw,
                                            int k0, int n0, int N, int tid) {
#pragma unroll
    for (int j = 0; j < 16; ++j) {
        int i = tid + j * 256;
        int kk = i >> 7, c = i & 127;
        int gc = n0 + c;
        r[j] = (gc < N) ? Bw[(size_t)(k0 + kk) * N + gc] : 0.0f;
    }
}

__device__ __forceinline__ void store_B_lds(__bf16* sB, const float* r, int tid) {
#pragma unroll
    for (int j = 0; j < 16; ++j) {
        int i = tid + j * 256;
        int kk = i >> 7, c = i & 127;
        sB[c * LDB + kk] = f2bf(r[j]);   // K-transposed
    }
}

__global__ __launch_bounds__(256)
void gemm_bf16_k(const __bf16* __restrict__ A, const float* __restrict__ Bw,
                 const float* __restrict__ bias, float* __restrict__ C,
                 int M, int N, int K) {
    constexpr int BM = 128, BN = 128, BK = 32;
    __shared__ __bf16 sA[2][BM * LDA];
    __shared__ __bf16 sB[2][BN * LDB];

    const int tid  = threadIdx.x;
    const int wave = tid >> 5, lane = tid & 31;
    const int wm = wave >> 2, wn = wave & 3;       // 2 x 4 wave grid
    const int lh = lane >> 4, lm = lane & 15;      // lane half / lane mod 16
    const int m0 = blockIdx.y * BM;
    const int n0 = blockIdx.x * BN;
    const __bf16* Abase = A + (size_t)m0 * K;

    const unsigned aLds[2] = { (unsigned)(uintptr_t)&sA[0][0],
                               (unsigned)(uintptr_t)&sA[1][0] };

    v8f acc[4][2] = {};
    float breg[16];
    const int nk = K / BK;   // 72

    // prologue: stage tiles for iteration 0 into buffer 0
    load_B_regs(breg, Bw, 0, n0, N, tid);
    stage_A_async(Abase, aLds[0], tid, K);
    store_B_lds(&sB[0][0], breg, tid);

    for (int it = 0; it < nk; ++it) {
        const int cur = it & 1, nxt = cur ^ 1;
        asm volatile("s_wait_asynccnt 0x0" ::: "memory");  // own A copies done
        __syncthreads();                                   // all waves' tiles visible

        // issue next tile's traffic before compute
        if (it + 1 < nk) {
            load_B_regs(breg, Bw, (it + 1) * BK, n0, N, tid);
            stage_A_async(Abase + (it + 1) * BK, aLds[nxt], tid, K);
        }

        // fragment loads (contiguous k = lh*16 + e, same for A and B) + 8 WMMAs
        v16bf af[4], bfm[2];
#pragma unroll
        for (int fm = 0; fm < 4; ++fm) {
            const __bf16* p = &sA[cur][(wm * 64 + fm * 16 + lm) * LDA];
#pragma unroll
            for (int e = 0; e < 8; ++e) af[fm][e]     = p[lh * 16 + e];
#pragma unroll
            for (int e = 0; e < 8; ++e) af[fm][8 + e] = p[lh * 16 + 8 + e];
        }
#pragma unroll
        for (int fn = 0; fn < 2; ++fn) {
            const __bf16* p = &sB[cur][(wn * 32 + fn * 16 + lm) * LDB];
#pragma unroll
            for (int e = 0; e < 16; ++e) bfm[fn][e] = p[lh * 16 + e];
        }
#pragma unroll
        for (int fm = 0; fm < 4; ++fm)
#pragma unroll
            for (int fn = 0; fn < 2; ++fn)
                acc[fm][fn] = __builtin_amdgcn_wmma_f32_16x16x32_bf16(
                    false, af[fm], false, bfm[fn], (short)0, acc[fm][fn], false, false);

        // convert + store next B tile after WMMAs are issued
        if (it + 1 < nk) store_B_lds(&sB[nxt][0], breg, tid);
    }

    // store D: VGPR v -> M = v + lh*8 ; N = lm
#pragma unroll
    for (int fn = 0; fn < 2; ++fn) {
        int col = n0 + wn * 32 + fn * 16 + lm;
        if (col >= N) continue;
        float bv = bias[col];
#pragma unroll
        for (int fm = 0; fm < 4; ++fm) {
#pragma unroll
            for (int v = 0; v < 8; ++v) {
                int rowg = m0 + wm * 64 + fm * 16 + v + lh * 8;
                C[(size_t)rowg * N + col] = acc[fm][fn][v] + bv;
            }
        }
    }
}

// ---------------------------------------------------------------------------
// Kernel 3: exact gelu + layernorm over rows of 2304, write bf16
// ---------------------------------------------------------------------------
__global__ __launch_bounds__(256)
void gelu_ln_k(const float* __restrict__ in, const float* __restrict__ g,
               const float* __restrict__ be, __bf16* __restrict__ out) {
    __shared__ float ws1[8], ws2[8];
    const int row  = blockIdx.x;
    const int tid  = threadIdx.x;
    const int wave = tid >> 5, lane = tid & 31;
    const float* p = in + (size_t)row * H3;

    float vals[9];
    float s = 0.f, s2 = 0.f;
#pragma unroll
    for (int j = 0; j < 9; ++j) {
        int i = tid + j * 256;
        float v = p[i];
        float gel = 0.5f * v * (1.0f + erff(v * 0.70710678118654752f));
        vals[j] = gel;
        s += gel;
        s2 += gel * gel;
    }
#pragma unroll
    for (int off = 16; off > 0; off >>= 1) {
        s  += __shfl_xor(s,  off, 32);
        s2 += __shfl_xor(s2, off, 32);
    }
    if (lane == 0) { ws1[wave] = s; ws2[wave] = s2; }
    __syncthreads();
    float ts = 0.f, ts2 = 0.f;
#pragma unroll
    for (int i = 0; i < 8; ++i) { ts += ws1[i]; ts2 += ws2[i]; }
    const float mu  = ts * (1.0f / H3);
    float var = ts2 * (1.0f / H3) - mu * mu;
    var = var < 0.f ? 0.f : var;
    const float inv = rsqrtf(var + 1e-5f);

    __bf16* o = out + (size_t)row * H3;
#pragma unroll
    for (int j = 0; j < 9; ++j) {
        int i = tid + j * 256;
        o[i] = f2bf((vals[j] - mu) * inv * g[i] + be[i]);
    }
}

// ---------------------------------------------------------------------------
extern "C" void kernel_launch(void* const* d_in, const int* in_sizes, int n_in,
                              void* d_out, int out_size, void* d_ws, size_t ws_size,
                              hipStream_t stream) {
    const float*         x    = (const float*)d_in[0];
    const int*           li   = (const int*)d_in[1];
    const int*           ri   = (const int*)d_in[2];
    const unsigned char* mask = (const unsigned char*)d_in[3];
    const float*         pos  = (const float*)d_in[4];
    const float*         W1   = (const float*)d_in[5];
    const float*         b1   = (const float*)d_in[6];
    const float*         g1   = (const float*)d_in[7];
    const float*         be1  = (const float*)d_in[8];
    const float*         W2   = (const float*)d_in[9];
    const float*         b2   = (const float*)d_in[10];
    const float*         g2   = (const float*)d_in[11];
    const float*         be2  = (const float*)d_in[12];
    const float*         Wout = (const float*)d_in[13];
    const float*         bout = (const float*)d_in[14];
    float*               out  = (float*)d_out;

    const size_t Mtok = (size_t)Bsz * Ssz;          // 4096
    char* ws = (char*)d_ws;
    __bf16* A0 = (__bf16*)ws;  ws += Mtok * H3 * sizeof(__bf16);   // concat  (bf16)
    __bf16* A1 = (__bf16*)ws;  ws += Mtok * H3 * sizeof(__bf16);   // h1 post-LN
    __bf16* A2 = (__bf16*)ws;  ws += Mtok * H3 * sizeof(__bf16);   // h2 post-LN
    float*  P1 = (float*)ws;                                       // pre-activation f32

    dim3 blk(256);
    // 1) concat
    build_concat_k<<<dim3((unsigned)Mtok), blk, 0, stream>>>(x, li, ri, mask, pos, A0);
    // 2) h1_pre = concat @ W1 + b1
    gemm_bf16_k<<<dim3(H3 / 128, (unsigned)(Mtok / 128)), blk, 0, stream>>>(
        A0, W1, b1, P1, (int)Mtok, H3, H3);
    // 3) h1 = LN(gelu(h1_pre))
    gelu_ln_k<<<dim3((unsigned)Mtok), blk, 0, stream>>>(P1, g1, be1, A1);
    // 4) h2_pre = h1 @ W2 + b2
    gemm_bf16_k<<<dim3(H3 / 128, (unsigned)(Mtok / 128)), blk, 0, stream>>>(
        A1, W2, b2, P1, (int)Mtok, H3, H3);
    // 5) h2 = LN(gelu(h2_pre))
    gelu_ln_k<<<dim3((unsigned)Mtok), blk, 0, stream>>>(P1, g2, be2, A2);
    // 6) out = h2 @ Wout + bout   (N = 30522, edge tiles guarded)
    gemm_bf16_k<<<dim3((Vsz + 127) / 128, (unsigned)(Mtok / 128)), blk, 0, stream>>>(
        A2, Wout, bout, out, (int)Mtok, Vsz, H3);
}
